// MultiHeadAttention_19945828122843
// MI455X (gfx1250) — matrix operand order repaired and checked by reading, not compiled, not run on hardware
//
#include <hip/hip_runtime.h>
#include <math.h>

// ---------------- problem constants ----------------
#define B_   16
#define LK_  2048
#define E_   2304
#define H_   8
#define D_   288          // E_/H_
#define TOK  32           // key tokens per block (chunk)
#define NCHUNK (LK_/TOK)  // 64
#define ROWB 2312         // padded LDS row stride in bf16 elems (2304+8) -> bank-conflict-free
#define KITER (E_/32)     // 72 wmma k-steps
#define NTW  (D_/16)      // 18 N-tiles per wave (= per head)
static constexpr float SCALE_ = 0.05892556509887896f; // 288^-0.5

// ---------------- types ----------------
typedef __bf16 bf16;
typedef bf16  v16bf __attribute__((ext_vector_type(16)));
typedef bf16  v8bf  __attribute__((ext_vector_type(8)));
typedef bf16  v4bf  __attribute__((ext_vector_type(4)));
typedef float v8f   __attribute__((ext_vector_type(8)));

union BF16x16 { v16bf v; v8bf h[2]; };

#define ABYTES (TOK * ROWB * 2)
#define SMEM_BYTES (ABYTES + 2 * H_ * TOK * 4)

// ---------------- fp32 -> bf16 weight pre-convert ----------------
__global__ void conv_bf16_kernel(const float* __restrict__ src, bf16* __restrict__ dst, int n4) {
  int i = blockIdx.x * blockDim.x + threadIdx.x;
  if (i >= n4) return;
  float4 f = ((const float4*)src)[i];
  v4bf o; o[0] = (bf16)f.x; o[1] = (bf16)f.y; o[2] = (bf16)f.z; o[3] = (bf16)f.w;
  ((v4bf*)dst)[i] = o;
}

// ---------------- q projection: q = query @ Wq.T + bq ----------------
__global__ void qproj_kernel(const float* __restrict__ query, const float* __restrict__ Wq,
                             const float* __restrict__ bq, float* __restrict__ qws) {
  int e = blockIdx.x * blockDim.x + threadIdx.x;
  if (e >= E_) return;
  float acc[B_];
#pragma unroll
  for (int b = 0; b < B_; ++b) acc[b] = 0.f;
  const float* wrow = Wq + (size_t)e * E_;
  for (int k = 0; k < E_; ++k) {
    float wv = wrow[k];
#pragma unroll
    for (int b = 0; b < B_; ++b) acc[b] += query[b * E_ + k] * wv;
  }
  float bias = bq[e];
#pragma unroll
  for (int b = 0; b < B_; ++b) qws[b * E_ + e] = acc[b] + bias;
}

// ---------------- shared helpers for the fused kernel ----------------
// stage a [32 x 2304] fp32 chunk (row stride E_) into LDS as bf16 with padded rows
__device__ __forceinline__ void stage32(bf16* __restrict__ bufA, const float* __restrict__ src,
                                        int tid) {
  for (int i = tid; i < TOK * (E_ / 4); i += 256) {
    int t  = i / (E_ / 4);
    int c4 = (i % (E_ / 4)) * 4;
    float4 f = *(const float4*)(src + (size_t)t * E_ + c4);
    v4bf o; o[0] = (bf16)f.x; o[1] = (bf16)f.y; o[2] = (bf16)f.z; o[3] = (bf16)f.w;
    *(v4bf*)(bufA + t * ROWB + c4) = o;
  }
}

// one 16x16 output tile pair (two M-subtiles of 16 tokens) accumulated over K=2304.
// A layout (16-bit 16x32): lane lm holds row M=lm; half hl selects K sub-blocks
//   elems 0..7 -> K = k0 + hl*8 + j ; elems 8..15 -> K = k0 + 16 + hl*8 + j
// B layout (16-bit 32x16): lane lm holds col N; same K mapping as A
// (plain loop: the compiler's own schedule -- loads for step kk+1 issued right after
//  step kk's WMMAs -- chosen over manual rotating buffers, which caused a
//  v_dual_mov copy storm)
__device__ __forceinline__ void gemm72(const bf16* __restrict__ wrow,
                                       const bf16* __restrict__ a_row0,
                                       const bf16* __restrict__ a_row1,
                                       int hl, v8f& acc0, v8f& acc1) {
#pragma unroll 4
  for (int kk = 0; kk < KITER; ++kk) {
    const int k0 = kk * 32;
    BF16x16 bu;
    bu.h[0] = *(const v8bf*)(wrow + k0 + hl * 8);
    bu.h[1] = *(const v8bf*)(wrow + k0 + 16 + hl * 8);
    v16bf bfr = bu.v;
    BF16x16 a0, a1;
    a0.h[0] = *(const v8bf*)(a_row0 + k0 + hl * 8);
    a0.h[1] = *(const v8bf*)(a_row0 + k0 + 16 + hl * 8);
    a1.h[0] = *(const v8bf*)(a_row1 + k0 + hl * 8);
    a1.h[1] = *(const v8bf*)(a_row1 + k0 + 16 + hl * 8);
    acc0 = __builtin_amdgcn_wmma_f32_16x16x32_bf16(false, a0.v, false, bfr, (short)0, acc0,
                                                   false, false);
    acc1 = __builtin_amdgcn_wmma_f32_16x16x32_bf16(false, a1.v, false, bfr, (short)0, acc1,
                                                   false, false);
  }
}

// ---------------- fused K/V projection + flash-decode attention chunk ----------------
__global__ void __launch_bounds__(256)
fused_kv_attn(const float* __restrict__ key, const float* __restrict__ value,
              const int* __restrict__ mask, const float* __restrict__ relb,
              const bf16* __restrict__ wkbf, const bf16* __restrict__ wvbf,
              const float* __restrict__ bk, const float* __restrict__ bv,
              const float* __restrict__ qws,
              float* __restrict__ Opart, float* __restrict__ mws, float* __restrict__ lws) {
  extern __shared__ char smem[];
  bf16*  bufA  = (bf16*)smem;
  float* s_arr = (float*)(smem + ABYTES);
  float* p_arr = s_arr + H_ * TOK;

  const int chunk = blockIdx.x;
  const int b     = blockIdx.y;
  const int tid   = threadIdx.x;
  const int w  = tid >> 5;          // wave == head
  const int l  = tid & 31;
  const int hl = l >> 4;            // lane half
  const int lm = l & 15;
  const int tok0 = chunk * TOK;

  // phase 0: stage raw key chunk into LDS (bf16)
  stage32(bufA, key + ((size_t)b * LK_ + tok0) * E_, tid);
  __syncthreads();

  const bf16* a_row0 = bufA + lm * ROWB;          // tokens 0..15  (A row = lm)
  const bf16* a_row1 = bufA + (16 + lm) * ROWB;   // tokens 16..31

  // phase 1: K projection for this head's 288 features; fold q.k scores straight
  // out of the accumulators. sacc[j]: j<8 -> token hl*8+j ; j>=8 -> token 16+hl*8+(j-8)
  float sacc[16];
#pragma unroll
  for (int j = 0; j < 16; ++j) sacc[j] = 0.f;

  for (int nt = 0; nt < NTW; ++nt) {
    const int n = w * D_ + nt * 16 + lm;   // global output-feature index (lane's C column)
    v8f acc0 = {}; v8f acc1 = {};
    gemm72(wkbf + (size_t)n * E_, a_row0, a_row1, hl, acc0, acc1);
    const float bkv = bk[n];
    const float qv  = qws[b * E_ + n];
#pragma unroll
    for (int r = 0; r < 8; ++r) {
      sacc[r]     += qv * (acc0[r] + bkv);
      sacc[8 + r] += qv * (acc1[r] + bkv);
    }
  }
  // reduce partial dot-products across the 16 lanes of each half (full 288-d sum)
#pragma unroll
  for (int offd = 1; offd < 16; offd <<= 1) {
#pragma unroll
    for (int j = 0; j < 16; ++j) sacc[j] += __shfl_xor(sacc[j], offd, 16);
  }
  if (lm == 0) {
#pragma unroll
    for (int j = 0; j < 16; ++j) {
      int tok = (j < 8) ? (hl * 8 + j) : (16 + hl * 8 + (j - 8));
      s_arr[w * TOK + tok] = sacc[j];
    }
  }
  __syncthreads();  // all waves done with bufA + s_arr visible

  // phase 2: local (chunk) softmax per head; lane == token
  {
    float s  = s_arr[w * TOK + l];
    int   mk = mask[b * LK_ + tok0 + l];
    float sv = (mk == 0) ? -1.0e30f : (s * SCALE_ + relb[b * H_ + w]);
    float mloc = sv;
#pragma unroll
    for (int offd = 1; offd < 32; offd <<= 1) mloc = fmaxf(mloc, __shfl_xor(mloc, offd, 32));
    float p  = (sv <= -1.0e29f) ? 0.f : __expf(sv - mloc);
    float ls = p;
#pragma unroll
    for (int offd = 1; offd < 32; offd <<= 1) ls += __shfl_xor(ls, offd, 32);
    p_arr[w * TOK + l] = p;
    if (l == 0) {
      mws[(b * H_ + w) * NCHUNK + chunk] = mloc;
      lws[(b * H_ + w) * NCHUNK + chunk] = ls;
    }
  }

  // phase 3: stage raw value chunk (bufA reuse) and do V projection; contract
  // attention weights straight out of the accumulators -> partial head output.
  stage32(bufA, value + ((size_t)b * LK_ + tok0) * E_, tid);
  __syncthreads();

  float pv[16];
#pragma unroll
  for (int j = 0; j < 16; ++j) {
    int tok = (j < 8) ? (hl * 8 + j) : (16 + hl * 8 + (j - 8));
    pv[j] = p_arr[w * TOK + tok];
  }

  for (int nt = 0; nt < NTW; ++nt) {
    const int n = w * D_ + nt * 16 + lm;
    v8f acc0 = {}; v8f acc1 = {};
    gemm72(wvbf + (size_t)n * E_, a_row0, a_row1, hl, acc0, acc1);
    const float bvv = bv[n];
    float od = 0.f;
#pragma unroll
    for (int r = 0; r < 8; ++r)
      od += pv[r] * (acc0[r] + bvv) + pv[8 + r] * (acc1[r] + bvv);
    od += __shfl_xor(od, 16, 32);   // add other half (same feature column)
    if (hl == 0)
      Opart[(((size_t)(b * H_ + w)) * NCHUNK + chunk) * D_ + nt * 16 + lm] = od;
  }
}

// ---------------- combine chunk partials (flash-decode reduction) ----------------
__global__ void rdx_kernel(const float* __restrict__ Opart, const float* __restrict__ mws,
                           const float* __restrict__ lws, float* __restrict__ attn) {
  int bh = blockIdx.x;       // B_*H_ blocks
  int d  = threadIdx.x;      // 288 threads
  float M = -1.0e30f;
  for (int c = 0; c < NCHUNK; ++c) M = fmaxf(M, mws[bh * NCHUNK + c]);
  float L = 0.f, o = 0.f;
  for (int c = 0; c < NCHUNK; ++c) {
    float wgt = __expf(mws[bh * NCHUNK + c] - M);
    L += lws[bh * NCHUNK + c] * wgt;
    o += wgt * Opart[((size_t)bh * NCHUNK + c) * D_ + d];
  }
  int b = bh / H_, h = bh % H_;
  attn[b * E_ + h * D_ + d] = o / fmaxf(L, 1e-30f);
}

// ---------------- output projection (WMMA, M = 16 = batch) ----------------
__global__ void __launch_bounds__(256)
oproj_kernel(const float* __restrict__ attn, const bf16* __restrict__ wobf,
             const float* __restrict__ bo, float* __restrict__ out) {
  int tid = threadIdx.x;
  int w = tid >> 5, l = tid & 31, hl = l >> 4, lm = l & 15;
  int ntg = blockIdx.x * 8 + w;           // 0..143
  int n   = ntg * 16 + lm;                // output feature column
  const bf16*  wrow = wobf + (size_t)n * E_;
  const float* arow = attn + (size_t)lm * E_;   // A row = batch index lm
  v8f acc = {};
#pragma unroll 4
  for (int kk = 0; kk < KITER; ++kk) {
    const int k0 = kk * 32;
    BF16x16 bu;
    bu.h[0] = *(const v8bf*)(wrow + k0 + hl * 8);
    bu.h[1] = *(const v8bf*)(wrow + k0 + 16 + hl * 8);
    v16bf bfr = bu.v;
    BF16x16 a;
    float4 f0 = *(const float4*)(arow + k0 + hl * 8);
    float4 f1 = *(const float4*)(arow + k0 + hl * 8 + 4);
    float4 f2 = *(const float4*)(arow + k0 + 16 + hl * 8);
    float4 f3 = *(const float4*)(arow + k0 + 16 + hl * 8 + 4);
    a.h[0][0] = (bf16)f0.x; a.h[0][1] = (bf16)f0.y; a.h[0][2] = (bf16)f0.z; a.h[0][3] = (bf16)f0.w;
    a.h[0][4] = (bf16)f1.x; a.h[0][5] = (bf16)f1.y; a.h[0][6] = (bf16)f1.z; a.h[0][7] = (bf16)f1.w;
    a.h[1][0] = (bf16)f2.x; a.h[1][1] = (bf16)f2.y; a.h[1][2] = (bf16)f2.z; a.h[1][3] = (bf16)f2.w;
    a.h[1][4] = (bf16)f3.x; a.h[1][5] = (bf16)f3.y; a.h[1][6] = (bf16)f3.z; a.h[1][7] = (bf16)f3.w;
    acc = __builtin_amdgcn_wmma_f32_16x16x32_bf16(false, a.v, false, bfr, (short)0, acc,
                                                  false, false);
  }
  float bon = bo[n];
#pragma unroll
  for (int r = 0; r < 8; ++r)
    out[(size_t)(r + 8 * hl) * E_ + n] = acc[r] + bon;   // row = batch, col = n
}

// ---------------- host launcher ----------------
extern "C" void kernel_launch(void* const* d_in, const int* in_sizes, int n_in,
                              void* d_out, int out_size, void* d_ws, size_t ws_size,
                              hipStream_t stream) {
  (void)in_sizes; (void)n_in; (void)out_size; (void)ws_size;
  const float* query = (const float*)d_in[0];
  const float* key   = (const float*)d_in[1];
  const float* value = (const float*)d_in[2];
  const int*   mask  = (const int*)  d_in[3];
  const float* Wq = (const float*)d_in[4];
  const float* bq = (const float*)d_in[5];
  const float* Wk = (const float*)d_in[6];
  const float* bk = (const float*)d_in[7];
  const float* Wv = (const float*)d_in[8];
  const float* bv = (const float*)d_in[9];
  const float* Wo = (const float*)d_in[10];
  const float* bo = (const float*)d_in[11];
  const float* rb = (const float*)d_in[12];
  float* out = (float*)d_out;

  // workspace carve-up
  char* ws = (char*)d_ws;
  auto al = [](size_t x) { return (x + 255) & ~(size_t)255; };
  size_t off = 0;
  bf16* wkbf = (bf16*)(ws + off); off = al(off + (size_t)E_ * E_ * sizeof(bf16));
  bf16* wvbf = (bf16*)(ws + off); off = al(off + (size_t)E_ * E_ * sizeof(bf16));
  bf16* wobf = (bf16*)(ws + off); off = al(off + (size_t)E_ * E_ * sizeof(bf16));
  float* qws   = (float*)(ws + off); off = al(off + (size_t)B_ * E_ * sizeof(float));
  float* Opart = (float*)(ws + off); off = al(off + (size_t)B_ * H_ * NCHUNK * D_ * sizeof(float));
  float* mws   = (float*)(ws + off); off = al(off + (size_t)B_ * H_ * NCHUNK * sizeof(float));
  float* lws   = (float*)(ws + off); off = al(off + (size_t)B_ * H_ * NCHUNK * sizeof(float));
  float* attnw = (float*)(ws + off); off = al(off + (size_t)B_ * E_ * sizeof(float));

  // 1) weights -> bf16 (stay L2-resident: 3 x 10.6 MB)
  const int n4 = E_ * E_ / 4;                 // 1,327,104
  conv_bf16_kernel<<<n4 / 256, 256, 0, stream>>>(Wk, wkbf, n4);
  conv_bf16_kernel<<<n4 / 256, 256, 0, stream>>>(Wv, wvbf, n4);
  conv_bf16_kernel<<<n4 / 256, 256, 0, stream>>>(Wo, wobf, n4);

  // 2) q projection (tiny)
  qproj_kernel<<<E_ / 256, 256, 0, stream>>>(query, Wq, bq, qws);

  // 3) fused K/V projection + chunked attention (bulk of the 0.7 TFLOP, WMMA)
  (void)hipFuncSetAttribute((const void*)fused_kv_attn,
                            hipFuncAttributeMaxDynamicSharedMemorySize, SMEM_BYTES);
  fused_kv_attn<<<dim3(NCHUNK, B_), 256, SMEM_BYTES, stream>>>(
      key, value, mask, rb, wkbf, wvbf, bk, bv, qws, Opart, mws, lws);

  // 4) combine chunk partials -> attn output [B, E]
  rdx_kernel<<<B_ * H_, D_, 0, stream>>>(Opart, mws, lws, attnw);

  // 5) output projection (WMMA, single 16-row M tile)
  oproj_kernel<<<(E_ / 16) / 8, 256, 0, stream>>>(attnw, wobf, bo, out);
}
